// SwinTransformerBlock3D_41446434406834
// MI455X (gfx1250) — compile-verified, matrix operand order checked by reading, not run on hardware
//
#include <hip/hip_runtime.h>
#include <math.h>

// ---------------------------------------------------------------------------
// Video Swin 3D block on gfx1250 (CDNA5, wave32, WMMA f16->f32).
// B=2,T=8,H=112,W=112,E=96, window (2,7,7) -> N=98 tokens, NH=3, hd=32.
// M = 200704 tokens total (divisible by 16).
// ---------------------------------------------------------------------------

typedef __attribute__((ext_vector_type(8)))  _Float16 v8h;
typedef __attribute__((ext_vector_type(16))) _Float16 v16h;
typedef __attribute__((ext_vector_type(8)))  float    v8f;

#define WMMA_F16(A,B,C) \
  __builtin_amdgcn_wmma_f32_16x16x32_f16(false,(A),false,(B),(short)0,(C),false,false)

static constexpr int MTOT = 200704;   // B*T*H*W
static constexpr int EDIM = 96;
static constexpr int NTOK = 98;
static constexpr int NWIN = 1024;     // windows per batch (4*16*16)

// Load one 16x32 f16 WMMA fragment slice for this lane from a row-major
// buffer. Caller passes pointer to (row, k0+kh) where kh = (lane>>4)*8.
// Lane layout (ISA 7.12.2): elems 0..7 = K(kh..kh+7), elems 8..15 = K(16+kh..).
static __device__ __forceinline__ v16h ldfrag(const _Float16* p) {
  v8h lo = *(const v8h*)(p);
  v8h hi = *(const v8h*)(p + 16);
  v16h r;
#pragma unroll
  for (int i = 0; i < 8; ++i) { r[i] = lo[i]; r[8 + i] = hi[i]; }
  return r;
}

// ---------------------------------------------------------------------------
// fp32 -> f16 weight cast
// ---------------------------------------------------------------------------
__global__ void cast_kernel(const float* __restrict__ s, _Float16* __restrict__ d, int n) {
  int i = blockIdx.x * blockDim.x + threadIdx.x;
  if (i < n) d[i] = (_Float16)s[i];
}

// ---------------------------------------------------------------------------
// LN1 + cyclic shift + window partition + cast to f16.
// Output row layout: (win, tok) with win = b*1024 + it*256 + ih*16 + iw,
// tok = (t'%2)*49 + (h'%7)*7 + (w'%7).
// ---------------------------------------------------------------------------
__global__ __launch_bounds__(128)
void ln1_shift_kernel(const float* __restrict__ x, const float* __restrict__ g,
                      const float* __restrict__ b, _Float16* __restrict__ xw) {
  int tok = blockIdx.x * blockDim.x + threadIdx.x;
  if (tok >= MTOT) return;
  int bb = tok / 100352;            // 8*112*112
  int r  = tok % 100352;
  int tp = r / 12544;               // 112*112
  int r2 = r % 12544;
  int hp = r2 / 112;
  int wp = r2 % 112;
  // shifted coord reads from original (t'+1, h'+3, w'+3) mod dims
  int ts = tp + 1;  if (ts >= 8)   ts -= 8;
  int hs = hp + 3;  if (hs >= 112) hs -= 112;
  int ws = wp + 3;  if (ws >= 112) ws -= 112;
  const float* px = x + ((((size_t)bb * 8 + ts) * 112 + hs) * 112 + ws) * EDIM;
  float v[EDIM];
  float mean = 0.f;
#pragma unroll
  for (int e = 0; e < EDIM; ++e) { v[e] = px[e]; mean += v[e]; }
  mean *= (1.f / EDIM);
  float m2 = 0.f;
#pragma unroll
  for (int e = 0; e < EDIM; ++e) { float d = v[e] - mean; m2 += d * d; }
  float inv = rsqrtf(m2 * (1.f / EDIM) + 1e-5f);
  int win = bb * NWIN + (tp >> 1) * 256 + (hp / 7) * 16 + (wp / 7);
  int n   = (tp & 1) * 49 + (hp % 7) * 7 + (wp % 7);
  _Float16* po = xw + ((size_t)win * NTOK + n) * EDIM;
#pragma unroll
  for (int e = 0; e < EDIM; ++e)
    po[e] = (_Float16)((v[e] - mean) * inv * g[e] + b[e]);
}

// ---------------------------------------------------------------------------
// LN2 (natural token order) -> f16
// ---------------------------------------------------------------------------
__global__ __launch_bounds__(128)
void ln2_kernel(const float* __restrict__ xo, const float* __restrict__ g,
                const float* __restrict__ b, _Float16* __restrict__ h2) {
  int tok = blockIdx.x * blockDim.x + threadIdx.x;
  if (tok >= MTOT) return;
  const float* px = xo + (size_t)tok * EDIM;
  float v[EDIM];
  float mean = 0.f;
#pragma unroll
  for (int e = 0; e < EDIM; ++e) { v[e] = px[e]; mean += v[e]; }
  mean *= (1.f / EDIM);
  float m2 = 0.f;
#pragma unroll
  for (int e = 0; e < EDIM; ++e) { float d = v[e] - mean; m2 += d * d; }
  float inv = rsqrtf(m2 * (1.f / EDIM) + 1e-5f);
  _Float16* po = h2 + (size_t)tok * EDIM;
#pragma unroll
  for (int e = 0; e < EDIM; ++e)
    po[e] = (_Float16)((v[e] - mean) * inv * g[e] + b[e]);
}

// ---------------------------------------------------------------------------
// Generic WMMA GEMM: C(M x N) = A(M x K) * W(N x K)^T + bias, fused epilogues.
// One wave per 16x48 output strip (3 f32 accumulators, A-frag reuse).
// EPI 0: f16 store, q-scale on cols < 96            (QKV)
// EPI 1: f16 store after exact GELU                 (MLP1)
// EPI 2: f32 window-reverse scatter + residual(x)   (proj)
// EPI 3: f32 store + residual(xo)                   (MLP2 -> d_out)
// ---------------------------------------------------------------------------
template <int KDIM, int EPI>
__global__ __launch_bounds__(128)
void gemm48_kernel(const _Float16* __restrict__ A, const _Float16* __restrict__ W,
                   const float* __restrict__ bias,
                   float* __restrict__ outF, _Float16* __restrict__ outH,
                   const float* __restrict__ resid,
                   int M, int N, float qscale) {
  const int wid  = (blockIdx.x * blockDim.x + threadIdx.x) >> 5;
  const int lane = threadIdx.x & 31;
  const int strips = N / 48;
  const int mt = wid / strips;
  const int n0 = (wid % strips) * 48;
  if (mt * 16 >= M) return;                   // wave-uniform
  const int lr = lane & 15;
  const int kh = (lane >> 4) << 3;            // 0 or 8

  v8f acc0 = {}, acc1 = {}, acc2 = {};
  const _Float16* pa = A + (size_t)(mt * 16 + lr) * KDIM + kh;
  const _Float16* w0 = W + (size_t)(n0 + lr) * KDIM + kh;
  const _Float16* w1 = w0 + (size_t)16 * KDIM;
  const _Float16* w2 = w0 + (size_t)32 * KDIM;
#pragma unroll
  for (int kb = 0; kb < KDIM / 32; ++kb) {
    v16h a  = ldfrag(pa + kb * 32);
    v16h b0 = ldfrag(w0 + kb * 32);
    v16h b1 = ldfrag(w1 + kb * 32);
    v16h b2 = ldfrag(w2 + kb * 32);
    acc0 = WMMA_F16(a, b0, acc0);
    acc1 = WMMA_F16(a, b1, acc1);
    acc2 = WMMA_F16(a, b2, acc2);
  }

  const int rbase = mt * 16 + ((lane >> 4) << 3);
#pragma unroll
  for (int j = 0; j < 3; ++j) {
    v8f acc = (j == 0) ? acc0 : ((j == 1) ? acc1 : acc2);
    const int col = n0 + j * 16 + lr;
    const float bv = bias[col];
#pragma unroll
    for (int r = 0; r < 8; ++r) {
      const int row = rbase + r;
      float v = acc[r] + bv;
      if (EPI == 0) {
        if (col < 96) v *= qscale;            // q rows get 1/sqrt(hd)
        outH[(size_t)row * N + col] = (_Float16)v;
      } else if (EPI == 1) {
        v = 0.5f * v * (1.0f + erff(v * 0.70710678118654752f));
        outH[(size_t)row * N + col] = (_Float16)v;
      } else if (EPI == 2) {
        // decode (window,tok) -> shifted coords -> natural coords (+SS roll)
        int widx = row / NTOK, tok = row % NTOK;
        int bb = widx >> 10, wi = widx & 1023;
        int it = wi >> 8, ih = (wi >> 4) & 15, iw = wi & 15;
        int tp = it * 2 + tok / 49;
        int rem = tok % 49;
        int hp = ih * 7 + rem / 7;
        int wp = iw * 7 + rem % 7;
        int t = tp + 1;   if (t >= 8)   t -= 8;
        int h = hp + 3;   if (h >= 112) h -= 112;
        int w = wp + 3;   if (w >= 112) w -= 112;
        size_t oidx = (((((size_t)bb * 8 + t) * 112 + h) * 112 + w)) * EDIM + col;
        outF[oidx] = v + resid[oidx];
      } else {                                 // EPI == 3
        size_t oidx = (size_t)row * N + col;
        outF[oidx] = v + resid[oidx];
      }
    }
  }
}

// ---------------------------------------------------------------------------
// Attention: one workgroup (4 waves) per (window, head).
// sS is pre-initialized with rel-pos bias + shift mask (coalesced, branch-light,
// before the tile loop), so the WMMA epilogue is a branch-free LDS +=.
// S = QK^T (7x7 tiles, K=32 -> one WMMA each), fp32 softmax in LDS,
// O = P*V (7x2 tiles, K=128 padded with zeros).
// ---------------------------------------------------------------------------
static constexpr int SS_BYTES  = 112 * 112 * 4;   // fp32 scores (stride 112)
static constexpr int SP_BYTES  = 112 * 128 * 2;   // f16 probs   (stride 128)
static constexpr int SQ_BYTES  = 112 * 32 * 2;
static constexpr int SK_BYTES  = 112 * 32 * 2;
static constexpr int SVT_BYTES = 32 * 128 * 2;    // V transposed, cols padded
static constexpr int ATTN_SMEM = SS_BYTES + SP_BYTES + SQ_BYTES + SK_BYTES + SVT_BYTES;

__global__ __launch_bounds__(128)
void attn_kernel(const _Float16* __restrict__ qkv,   // (M, 288)
                 const float* __restrict__ rpb,      // (507, 3)
                 const int* __restrict__ relidx,     // (98, 98)
                 const float* __restrict__ maskm,    // (1024, 98, 98)
                 _Float16* __restrict__ o)           // (M, 96)
{
  extern __shared__ char smem[];
  float*    sS  = (float*)smem;
  _Float16* sP  = (_Float16*)(smem + SS_BYTES);
  _Float16* sQ  = (_Float16*)(smem + SS_BYTES + SP_BYTES);
  _Float16* sK  = (_Float16*)(smem + SS_BYTES + SP_BYTES + SQ_BYTES);
  _Float16* sVt = (_Float16*)(smem + SS_BYTES + SP_BYTES + SQ_BYTES + SK_BYTES);

  const int win = blockIdx.x / 3;
  const int hd  = blockIdx.x % 3;
  const int tid = threadIdx.x;
  const int lane = tid & 31;
  const int wv   = tid >> 5;
  const int lr = lane & 15;
  const int kh = (lane >> 4) << 3;

  // Stage Q,K (112x32, zero-padded rows) and V^T (32x128, zero-padded cols).
  for (int idx = tid; idx < 112 * 32; idx += 128) {
    int n = idx >> 5, d = idx & 31;
    _Float16 qv = (_Float16)0, kv = (_Float16)0, vv = (_Float16)0;
    if (n < NTOK) {
      const _Float16* p = qkv + ((size_t)win * NTOK + n) * 288 + hd * 32 + d;
      qv = p[0]; kv = p[96]; vv = p[192];
    }
    sQ[idx] = qv;
    sK[idx] = kv;
    sVt[d * 128 + n] = vv;
  }
  for (int idx = tid; idx < 32 * 16; idx += 128)        // V^T cols 112..127
    sVt[(idx >> 4) * 128 + 112 + (idx & 15)] = (_Float16)0;

  // Pre-fill scores with bias + mask (padding -> -1e30), coalesced in m.
  {
    const float* mbase = maskm + (size_t)(win & (NWIN - 1)) * (NTOK * NTOK);
    for (int idx = tid; idx < 112 * 112; idx += 128) {
      int n = idx / 112, m = idx % 112;
      float sv = -1e30f;
      if (n < NTOK && m < NTOK) {
        int nm = n * NTOK + m;
        sv = rpb[relidx[nm] * 3 + hd] + mbase[nm];
      }
      sS[idx] = sv;
    }
  }
  __syncthreads();

  // S += Q K^T (49 tiles split across 4 waves); epilogue is branch-free.
  for (int t = wv; t < 49; t += 4) {
    int ti = t / 7, tj = t % 7;
    v16h a = ldfrag(sQ + (ti * 16 + lr) * 32 + kh);
    v16h b = ldfrag(sK + (tj * 16 + lr) * 32 + kh);
    v8f acc = {};
    acc = WMMA_F16(a, b, acc);
    float* sc = sS + (ti * 16 + ((lane >> 4) << 3)) * 112 + tj * 16 + lr;
#pragma unroll
    for (int r = 0; r < 8; ++r)
      sc[r * 112] += acc[r];
  }
  __syncthreads();

  // Row softmax (fp32) -> f16 probabilities, cols padded to 128 with zeros.
  if (tid < NTOK) {
    const float* row = sS + tid * 112;
    float mx = -1e30f;
    for (int m = 0; m < NTOK; ++m) mx = fmaxf(mx, row[m]);
    float s = 0.f;
    for (int m = 0; m < NTOK; ++m) s += __expf(row[m] - mx);
    float inv = 1.f / s;
    _Float16* pr = sP + tid * 128;
    for (int m = 0; m < NTOK; ++m) pr[m] = (_Float16)(__expf(row[m] - mx) * inv);
    for (int m = NTOK; m < 128; ++m) pr[m] = (_Float16)0;
  } else if (tid < 112) {
    _Float16* pr = sP + tid * 128;
    for (int m = 0; m < 128; ++m) pr[m] = (_Float16)0;
  }
  __syncthreads();

  // O = P V : 7 row-tiles x 2 d-tiles, K = 128 (4 WMMA steps)
  for (int t = wv; t < 14; t += 4) {
    int ti = t >> 1, tdj = t & 1;
    v8f acc = {};
#pragma unroll
    for (int kb = 0; kb < 4; ++kb) {
      v16h a = ldfrag(sP  + (ti  * 16 + lr) * 128 + kb * 32 + kh);
      v16h b = ldfrag(sVt + (tdj * 16 + lr) * 128 + kb * 32 + kh);
      acc = WMMA_F16(a, b, acc);
    }
    int dcol = tdj * 16 + lr;
#pragma unroll
    for (int r = 0; r < 8; ++r) {
      int nrow = ti * 16 + ((lane >> 4) << 3) + r;
      if (nrow < NTOK)
        o[((size_t)win * NTOK + nrow) * EDIM + hd * 32 + dcol] = (_Float16)acc[r];
    }
  }
}

// ---------------------------------------------------------------------------
// Launch
// ---------------------------------------------------------------------------
extern "C" void kernel_launch(void* const* d_in, const int* in_sizes, int n_in,
                              void* d_out, int out_size, void* d_ws, size_t ws_size,
                              hipStream_t stream) {
  const float* x      = (const float*)d_in[0];
  const float* maskm  = (const float*)d_in[1];
  const int*   relidx = (const int*)d_in[2];
  const float* g1     = (const float*)d_in[3];
  const float* b1     = (const float*)d_in[4];
  const float* w_qkv  = (const float*)d_in[5];
  const float* b_qkv  = (const float*)d_in[6];
  const float* rpb    = (const float*)d_in[7];
  const float* w_proj = (const float*)d_in[8];
  const float* b_proj = (const float*)d_in[9];
  const float* g2     = (const float*)d_in[10];
  const float* b2     = (const float*)d_in[11];
  const float* w_mlp1 = (const float*)d_in[12];
  const float* b_mlp1 = (const float*)d_in[13];
  const float* w_mlp2 = (const float*)d_in[14];
  const float* b_mlp2 = (const float*)d_in[15];
  float* out = (float*)d_out;

  char* ws = (char*)d_ws;
  size_t off = 0;
  auto take = [&](size_t bytes) {
    char* p = ws + off;
    off += (bytes + 255) & ~(size_t)255;
    return p;
  };
  _Float16* wqkvh  = (_Float16*)take((size_t)288 * 96 * 2);
  _Float16* wprojh = (_Float16*)take((size_t)96 * 96 * 2);
  _Float16* wm1h   = (_Float16*)take((size_t)384 * 96 * 2);
  _Float16* wm2h   = (_Float16*)take((size_t)96 * 384 * 2);
  _Float16* xwh    = (_Float16*)take((size_t)MTOT * 96 * 2);
  _Float16* qkvh   = (_Float16*)take((size_t)MTOT * 288 * 2);
  _Float16* oh     = (_Float16*)take((size_t)MTOT * 96 * 2);
  float*    xof    = (float*)   take((size_t)MTOT * 96 * 4);
  _Float16* h2h    = (_Float16*)take((size_t)MTOT * 96 * 2);
  _Float16* hidh   = (_Float16*)take((size_t)MTOT * 384 * 2);

  // weight casts
  cast_kernel<<<(27648 + 255) / 256, 256, 0, stream>>>(w_qkv,  wqkvh,  27648);
  cast_kernel<<<(9216  + 255) / 256, 256, 0, stream>>>(w_proj, wprojh, 9216);
  cast_kernel<<<(36864 + 255) / 256, 256, 0, stream>>>(w_mlp1, wm1h,   36864);
  cast_kernel<<<(36864 + 255) / 256, 256, 0, stream>>>(w_mlp2, wm2h,   36864);

  // LN1 + shift + window partition
  ln1_shift_kernel<<<MTOT / 128, 128, 0, stream>>>(x, g1, b1, xwh);

  const int mtiles = MTOT / 16;                      // 12544
  const float qscale = 0.17677669529663689f;         // 32^-0.5

  // QKV: (M,96)x(288,96)^T, q-scale fused
  gemm48_kernel<96, 0><<<mtiles * (288 / 48) / 4, 128, 0, stream>>>(
      xwh, wqkvh, b_qkv, nullptr, qkvh, nullptr, MTOT, 288, qscale);

  // attention per (window, head)
  attn_kernel<<<2048 * 3, 128, ATTN_SMEM, stream>>>(qkvh, rpb, relidx, maskm, oh);

  // proj + window-reverse + inverse shift + residual(x) -> xo (natural order)
  gemm48_kernel<96, 2><<<mtiles * (96 / 48) / 4, 128, 0, stream>>>(
      oh, wprojh, b_proj, xof, nullptr, x, MTOT, 96, 1.f);

  // LN2
  ln2_kernel<<<MTOT / 128, 128, 0, stream>>>(xof, g2, b2, h2h);

  // MLP1 + GELU
  gemm48_kernel<96, 1><<<mtiles * (384 / 48) / 4, 128, 0, stream>>>(
      h2h, wm1h, b_mlp1, nullptr, hidh, nullptr, MTOT, 384, 1.f);

  // MLP2 + residual(xo) -> final fp32 output
  gemm48_kernel<384, 3><<<mtiles * (96 / 48) / 4, 128, 0, stream>>>(
      hidh, wm2h, b_mlp2, out, nullptr, xof, MTOT, 96, 1.f);
}
